// STNProjector3D_81862076661795
// MI455X (gfx1250) — matrix operand order repaired and checked by reading, not compiled, not run on hardware
//
#include <hip/hip_runtime.h>
#include <hip/hip_bf16.h>
#include <math.h>

typedef __attribute__((ext_vector_type(16))) _Float16 v16h;
typedef __attribute__((ext_vector_type(8)))  float    v8f;

// Problem constants
#define Bv    2
#define Vv    6
#define Cc    96
#define Dd    48
#define Hh    48
#define Ww    48
#define N48   2304            // 48*48
#define N96   9216            // 96*96
#define NPV   18432           // B * 96*96 columns per view
#define K1    864             // C*9
#define KC1   27              // K1/32
#define KC2   3               // 96/32
#define BN_CH 72              // NPV / 256 partial-sum chunks

// ---------------------------------------------------------------------------
// Kernel 1: pack conv weights into exact WMMA A-fragment order (f16).
// A-fragment layout used here (16x32 f16, contiguous k, not the ISA table order): lane L (half = L>>4, M = L&15),
// element j: klocal = j + 16*half.
// ap1[((mt*27+kc)*32+lane)*16+j],  k = kc*32+klocal, k = tap*96 + cin.
// ---------------------------------------------------------------------------
__global__ __launch_bounds__(256) void pack_weights_kernel(
    const float* __restrict__ w1, const float* __restrict__ w2,
    _Float16* __restrict__ ap1, _Float16* __restrict__ ap2) {
  int t = blockIdx.x * 256 + threadIdx.x;
  if (t < 82944) {            // 6*27*512
    int j    = t & 15;
    int lane = (t >> 4) & 31;
    int kc   = (t >> 9) % KC1;
    int mt   = t / (KC1 * 512);
    int half = lane >> 4;
    int klocal = j + 16 * half;
    int k   = kc * 32 + klocal;
    int tap = k / Cc, cin = k % Cc;
    int ky  = tap / 3, kx = tap % 3;
    int m   = mt * 16 + (lane & 15);
    ap1[t] = (_Float16)w1[((m * Cc + cin) * 3 + ky) * 3 + kx];
    return;
  }
  int u = t - 82944;
  if (u >= 0 && u < 9216) {   // 6*3*512
    int j    = u & 15;
    int lane = (u >> 4) & 31;
    int kc   = (u >> 9) % KC2;
    int mt   = u / (KC2 * 512);
    int half = lane >> 4;
    int klocal = j + 16 * half;
    int k = kc * 32 + klocal;
    int m = mt * 16 + (lane & 15);
    ap2[u] = (_Float16)w2[m * Cc + k];
  }
}

// ---------------------------------------------------------------------------
// Kernel 2: rotate volume + trilinear sample (zeros OOB) + mean over depth.
// vol[b][c][z][y][x] = voxel_tokens[b][(z*H+y)*W+x][c]  (channel-last!)
// One block per (v,b,oy,ox); threadIdx.x = channel -> all 8 corner gathers
// are lane-contiguous (stride-1 in c). Volume (85 MB) is L2-resident.
// feat48 layout: [((v*2+b)*2304 + oy*48+ox)*96 + c]  (NHWC)
// ---------------------------------------------------------------------------
__global__ __launch_bounds__(96) void proj_kernel(
    const float* __restrict__ vox, const float* __restrict__ angles,
    float* __restrict__ feat48) {
  int blk = blockIdx.x;                // (v*2+b)*2304 + pix
  int c   = threadIdx.x;
  int pix = blk % N48;
  int vb  = blk / N48;
  int b   = vb % Bv;
  int v   = vb / Bv;
  int oy  = pix / Ww, ox = pix % Ww;

  float theta = angles[v];
  float ct = cosf(theta), st = sinf(theta);
  float X = (ox + 0.5f) * (2.0f / Ww) - 1.0f;
  float Y = (oy + 0.5f) * (2.0f / Hh) - 1.0f;

  // y (gy = Y) is depth-invariant
  float iy  = ((Y + 1.0f) * Hh - 1.0f) * 0.5f;
  float y0f = floorf(iy);
  float ty  = iy - y0f;
  int   y0  = (int)y0f;

  const float* volb = vox + (size_t)b * (Dd * Hh * Ww) * Cc;
  float acc = 0.0f;

  for (int izo = 0; izo < Dd; ++izo) {
    float Z  = (izo + 0.5f) * (2.0f / Dd) - 1.0f;
    float gx =  ct * X + st * Z;
    float gz = -st * X + ct * Z;
    float ix = ((gx + 1.0f) * Ww - 1.0f) * 0.5f;
    float iz = ((gz + 1.0f) * Dd - 1.0f) * 0.5f;
    float x0f = floorf(ix), z0f = floorf(iz);
    float tx = ix - x0f, tz = iz - z0f;
    int x0 = (int)x0f, z0 = (int)z0f;
#pragma unroll
    for (int dz = 0; dz < 2; ++dz)
#pragma unroll
      for (int dy = 0; dy < 2; ++dy)
#pragma unroll
        for (int dx = 0; dx < 2; ++dx) {
          int xc = x0 + dx, yc = y0 + dy, zc = z0 + dz;
          bool inb = ((unsigned)xc < (unsigned)Ww) &
                     ((unsigned)yc < (unsigned)Hh) &
                     ((unsigned)zc < (unsigned)Dd);
          int xi = xc < 0 ? 0 : (xc > Ww - 1 ? Ww - 1 : xc);
          int yi = yc < 0 ? 0 : (yc > Hh - 1 ? Hh - 1 : yc);
          int zi = zc < 0 ? 0 : (zc > Dd - 1 ? Dd - 1 : zc);
          float w = (dx ? tx : 1.0f - tx) * (dy ? ty : 1.0f - ty) *
                    (dz ? tz : 1.0f - tz);
          float val = volb[((size_t)(zi * (Hh * Ww) + yi * Ww + xi)) * Cc + c];
          acc += val * (inb ? w : 0.0f);
        }
  }
  feat48[(size_t)blk * Cc + c] = acc * (1.0f / Dd);
}

// ---------------------------------------------------------------------------
// Kernel 3: bilinear resize 48x48 -> 96x96, write f16 NHWC for WMMA B-frags.
// feat_h layout: [((v*2+b)*9216 + oy*96+ox)*96 + c]
// ---------------------------------------------------------------------------
__global__ __launch_bounds__(96) void resize_kernel(
    const float* __restrict__ feat48, _Float16* __restrict__ feat_h) {
  int blk = blockIdx.x;               // (v*2+b)*9216 + pix
  int c   = threadIdx.x;
  int pix = blk % N96;
  int vb  = blk / N96;
  int oy  = pix / 96, ox = pix % 96;

  float sy = fmaxf(0.5f * (oy + 0.5f) - 0.5f, 0.0f);
  int   y0 = (int)floorf(sy); if (y0 > 47) y0 = 47;
  int   y1 = y0 + 1 > 47 ? 47 : y0 + 1;
  float ty = sy - (float)y0;

  float sx = fmaxf(0.5f * (ox + 0.5f) - 0.5f, 0.0f);
  int   x0 = (int)floorf(sx); if (x0 > 47) x0 = 47;
  int   x1 = x0 + 1 > 47 ? 47 : x0 + 1;
  float tx = sx - (float)x0;

  const float* inp = feat48 + (size_t)vb * N48 * Cc;
  float v00 = inp[((size_t)(y0 * 48 + x0)) * Cc + c];
  float v10 = inp[((size_t)(y1 * 48 + x0)) * Cc + c];
  float v01 = inp[((size_t)(y0 * 48 + x1)) * Cc + c];
  float v11 = inp[((size_t)(y1 * 48 + x1)) * Cc + c];
  float r0 = v00 * (1.0f - ty) + v10 * ty;
  float r1 = v01 * (1.0f - ty) + v11 * ty;
  float val = r0 * (1.0f - tx) + r1 * tx;
  feat_h[(size_t)blk * Cc + c] = (_Float16)val;
}

// ---------------------------------------------------------------------------
// Kernel 4: conv3x3 (SAME, no bias) as im2col GEMM with v_wmma_f32_16x16x32_f16.
// M = 96 c_out (6 tiles), K = 864 (27 chunks, fully unrolled so tap/dy/dx and
// channel offsets are compile-time), N: TWO 16-column tiles per wave so the
// 12 A-fragment loads per K-chunk feed 12 WMMAs (load:wmma = 1.33 vs 2.33).
// conv_out layout NHWC: [(v*18432 + n)*96 + m]
// ---------------------------------------------------------------------------
__global__ __launch_bounds__(128) void conv3_wmma_kernel(
    const _Float16* __restrict__ feat_h, const _Float16* __restrict__ ap1,
    float* __restrict__ conv_out) {
  int wave = threadIdx.x >> 5;
  int lane = threadIdx.x & 31;
  int half = lane >> 4;
  int col  = lane & 15;

  const int blkPerView = NPV / 128;   // 144 (4 waves x 32 columns)
  int v     = blockIdx.x / blkPerView;
  int nb    = (blockIdx.x % blkPerView) * 128 + wave * 32;
  int ncol0 = nb + col;
  int ncol1 = ncol0 + 16;

  int b0 = ncol0 / N96, p0 = ncol0 % N96;
  int b1 = ncol1 / N96, p1 = ncol1 % N96;
  int y0p = p0 / 96, x0p = p0 % 96;
  int y1p = p1 / 96, x1p = p1 % 96;

  // Base pointers: all subsequent offsets are compile-time immediates.
  const _Float16* bbase0 =
      feat_h + ((size_t)(v * NPV + b0 * N96 + y0p * 96 + x0p)) * Cc + half * 16;
  const _Float16* bbase1 =
      feat_h + ((size_t)(v * NPV + b1 * N96 + y1p * 96 + x1p)) * Cc + half * 16;
  const _Float16* abase = ap1 + (size_t)lane * 16;

  v8f acc0[6], acc1[6];
#pragma unroll
  for (int i = 0; i < 6; ++i) { acc0[i] = (v8f){}; acc1[i] = (v8f){}; }

#pragma unroll
  for (int kc = 0; kc < KC1; ++kc) {
    const int tap  = kc / 3;                 // ky*3+kx (compile-time)
    const int dy   = tap / 3 - 1;
    const int dx   = tap % 3 - 1;
    const int coff = (kc % 3) * 32;          // channel offset within tap
    const int boff = (dy * 96 + dx) * Cc + coff;

    int yy0 = y0p + dy, xx0 = x0p + dx;
    int yy1 = y1p + dy, xx1 = x1p + dx;

    v16h bfrag0 = {}, bfrag1 = {};
    if (((unsigned)yy0 < 96u) & ((unsigned)xx0 < 96u))
      bfrag0 = *(const v16h*)(bbase0 + boff);
    if (((unsigned)yy1 < 96u) & ((unsigned)xx1 < 96u))
      bfrag1 = *(const v16h*)(bbase1 + boff);

#pragma unroll
    for (int mt = 0; mt < 6; ++mt) {
      v16h afrag = *(const v16h*)(abase + mt * (KC1 * 512) + kc * 512);
      acc0[mt] = __builtin_amdgcn_wmma_f32_16x16x32_f16(
          false, afrag, false, bfrag0, (short)0, acc0[mt], false, false);
      acc1[mt] = __builtin_amdgcn_wmma_f32_16x16x32_f16(
          false, afrag, false, bfrag1, (short)0, acc1[mt], false, false);
    }
  }

  // C/D layout: VGPR r, lanes 0-15 -> M=r, lanes 16-31 -> M=8+r; N = lane&15
#pragma unroll
  for (int mt = 0; mt < 6; ++mt)
#pragma unroll
    for (int r = 0; r < 8; ++r) {
      int m = mt * 16 + half * 8 + r;
      conv_out[((size_t)(v * NPV + ncol0)) * Cc + m] = acc0[mt][r];
      conv_out[((size_t)(v * NPV + ncol1)) * Cc + m] = acc1[mt][r];
    }
}

// ---------------------------------------------------------------------------
// Kernel 5a: BN partial sums. threadIdx.x = channel -> fully coalesced reads
// (96 contiguous floats per iteration). 72 chunks of 256 columns per view.
// ---------------------------------------------------------------------------
__global__ __launch_bounds__(96) void bnpart_kernel(
    const float* __restrict__ conv_out, float* __restrict__ ps,
    float* __restrict__ ps2) {
  int blk = blockIdx.x;               // v*BN_CH + ch
  int m   = threadIdx.x;
  int ch  = blk % BN_CH;
  int v   = blk / BN_CH;
  const float* p = conv_out + ((size_t)(v * NPV + ch * 256)) * Cc + m;
  float s = 0.0f, s2 = 0.0f;
  for (int i = 0; i < 256; ++i) {
    float xv = p[(size_t)i * Cc];
    s += xv;
    s2 += xv * xv;
  }
  ps [(size_t)blk * Cc + m] = s;
  ps2[(size_t)blk * Cc + m] = s2;
}

// ---------------------------------------------------------------------------
// Kernel 5b: finalize mean / inv-std (deterministic serial partial reduce).
// ---------------------------------------------------------------------------
__global__ __launch_bounds__(96) void bnfinal_kernel(
    const float* __restrict__ ps, const float* __restrict__ ps2,
    float* __restrict__ mu, float* __restrict__ rs) {
  int vm = blockIdx.x * 96 + threadIdx.x;   // grid = Vv blocks of 96
  int m = vm % Cc, v = vm / Cc;
  float s = 0.0f, s2 = 0.0f;
  for (int ch = 0; ch < BN_CH; ++ch) {
    s  += ps [(size_t)(v * BN_CH + ch) * Cc + m];
    s2 += ps2[(size_t)(v * BN_CH + ch) * Cc + m];
  }
  float mean = s * (1.0f / NPV);
  float var  = s2 * (1.0f / NPV) - mean * mean;
  mu[vm] = mean;
  rs[vm] = rsqrtf(var + 1e-5f);
}

// ---------------------------------------------------------------------------
// Kernel 6: normalize + affine + exact GELU, write f16 NHWC (reuses feat_h).
// ---------------------------------------------------------------------------
__global__ __launch_bounds__(256) void bn_gelu_kernel(
    const float* __restrict__ conv_out, const float* __restrict__ mu,
    const float* __restrict__ rs, const float* __restrict__ gamma,
    const float* __restrict__ beta, _Float16* __restrict__ act_h) {
  size_t t = (size_t)blockIdx.x * 256 + threadIdx.x;
  const size_t total = (size_t)Vv * NPV * Cc;
  if (t >= total) return;
  int c = (int)(t % Cc);
  int q = (int)(t / Cc);              // v*18432 + n
  int v = q / NPV;
  float xv = conv_out[t];
  float xn = (xv - mu[v * Cc + c]) * rs[v * Cc + c];
  float yv = xn * gamma[c] + beta[c];
  float g  = 0.5f * yv * (1.0f + erff(yv * 0.70710678118654752f));
  act_h[t] = (_Float16)g;
}

// ---------------------------------------------------------------------------
// Kernel 7: conv1x1 GEMM (K=96, 3 chunks) + bias, two N-tiles per wave,
// write final (B,V,C,96,96).
// ---------------------------------------------------------------------------
__global__ __launch_bounds__(128) void conv1_wmma_kernel(
    const _Float16* __restrict__ act_h, const _Float16* __restrict__ ap2,
    const float* __restrict__ bias, float* __restrict__ out) {
  int wave = threadIdx.x >> 5;
  int lane = threadIdx.x & 31;
  int half = lane >> 4;
  int col  = lane & 15;

  const int blkPerView = NPV / 128;   // 144
  int v     = blockIdx.x / blkPerView;
  int nb    = (blockIdx.x % blkPerView) * 128 + wave * 32;
  int ncol0 = nb + col;
  int ncol1 = ncol0 + 16;
  int b0 = ncol0 / N96, p0 = ncol0 % N96;
  int b1 = ncol1 / N96, p1 = ncol1 % N96;

  const _Float16* bbase0 =
      act_h + ((size_t)(v * NPV + ncol0)) * Cc + half * 16;
  const _Float16* bbase1 =
      act_h + ((size_t)(v * NPV + ncol1)) * Cc + half * 16;
  const _Float16* abase = ap2 + (size_t)lane * 16;

  v8f acc0[6], acc1[6];
#pragma unroll
  for (int i = 0; i < 6; ++i) { acc0[i] = (v8f){}; acc1[i] = (v8f){}; }

#pragma unroll
  for (int kc = 0; kc < KC2; ++kc) {
    v16h bfrag0 = *(const v16h*)(bbase0 + kc * 32);
    v16h bfrag1 = *(const v16h*)(bbase1 + kc * 32);
#pragma unroll
    for (int mt = 0; mt < 6; ++mt) {
      v16h afrag = *(const v16h*)(abase + mt * (KC2 * 512) + kc * 512);
      acc0[mt] = __builtin_amdgcn_wmma_f32_16x16x32_f16(
          false, afrag, false, bfrag0, (short)0, acc0[mt], false, false);
      acc1[mt] = __builtin_amdgcn_wmma_f32_16x16x32_f16(
          false, afrag, false, bfrag1, (short)0, acc1[mt], false, false);
    }
  }

#pragma unroll
  for (int mt = 0; mt < 6; ++mt)
#pragma unroll
    for (int r = 0; r < 8; ++r) {
      int m = mt * 16 + half * 8 + r;
      // out shape (B, V, C, 96, 96)
      out[((size_t)((b0 * Vv + v) * Cc + m)) * N96 + p0] = acc0[mt][r] + bias[m];
      out[((size_t)((b1 * Vv + v) * Cc + m)) * N96 + p1] = acc1[mt][r] + bias[m];
    }
}

// ---------------------------------------------------------------------------
extern "C" void kernel_launch(void* const* d_in, const int* in_sizes, int n_in,
                              void* d_out, int out_size, void* d_ws,
                              size_t ws_size, hipStream_t stream) {
  const float* vox   = (const float*)d_in[0];  // (B, D*H*W, C)
  const float* ang   = (const float*)d_in[1];  // (V,)
  const float* w1    = (const float*)d_in[2];  // (C,C,3,3)
  const float* gamma = (const float*)d_in[3];  // (C,)
  const float* beta  = (const float*)d_in[4];  // (C,)
  const float* w2    = (const float*)d_in[5];  // (C,C,1,1)
  const float* b2    = (const float*)d_in[6];  // (C,)
  float* out = (float*)d_out;

  char* ws = (char*)d_ws;
  size_t off = 0;
  auto carve = [&](size_t bytes) -> void* {
    void* pp = ws + off;
    off = (off + bytes + 255) & ~(size_t)255;
    return pp;
  };
  float*    feat48 = (float*)   carve((size_t)Vv * Bv * N48 * Cc * 4);  // 10.6 MB
  _Float16* feat_h = (_Float16*)carve((size_t)Vv * NPV * Cc * 2);       // 21.2 MB
  _Float16* ap1    = (_Float16*)carve((size_t)6 * KC1 * 512 * 2);
  _Float16* ap2    = (_Float16*)carve((size_t)6 * KC2 * 512 * 2);
  float*    convo  = (float*)   carve((size_t)Vv * NPV * Cc * 4);       // 42.5 MB
  float*    ps     = (float*)   carve((size_t)Vv * BN_CH * Cc * 4);
  float*    ps2    = (float*)   carve((size_t)Vv * BN_CH * Cc * 4);
  float*    mu     = (float*)   carve((size_t)Vv * Cc * 4);
  float*    rs     = (float*)   carve((size_t)Vv * Cc * 4);

  pack_weights_kernel<<<360, 256, 0, stream>>>(w1, w2, ap1, ap2);
  proj_kernel<<<Vv * Bv * N48, 96, 0, stream>>>(vox, ang, feat48);
  resize_kernel<<<Vv * Bv * N96, 96, 0, stream>>>(feat48, feat_h);
  conv3_wmma_kernel<<<Vv * (NPV / 128), 128, 0, stream>>>(feat_h, ap1, convo);
  bnpart_kernel<<<Vv * BN_CH, 96, 0, stream>>>(convo, ps, ps2);
  bnfinal_kernel<<<Vv, 96, 0, stream>>>(ps, ps2, mu, rs);
  {
    size_t total = (size_t)Vv * NPV * Cc;
    bn_gelu_kernel<<<(unsigned)((total + 255) / 256), 256, 0, stream>>>(
        convo, mu, rs, gamma, beta, feat_h);   // reuse feat_h as activation
  }
  conv1_wmma_kernel<<<Vv * (NPV / 128), 128, 0, stream>>>(feat_h, ap2, b2, out);
}
